// HybridCSNN_2473901162851
// MI455X (gfx1250) — matrix operand order repaired and checked by reading, not compiled, not run on hardware
//
#include <hip/hip_runtime.h>
#include <hip/hip_bf16.h>

typedef __attribute__((ext_vector_type(16))) __bf16 v16bf;
typedef __attribute__((ext_vector_type(8)))  float  v8f;

#define B_    1024
#define H_IN  64
#define C1    16
#define H1    60            // 64 - 4
#define C2    32
#define H2    56            // 60 - 4
#define HP    28
#define NFEAT (C2*HP*HP)    // 25088
#define NSTEPS 25
#define NOUT  10
#define TILES_PER_IMG ((H2*H2)/16)  // 196
#define NCHUNK 13                    // K = 400 padded to 13*32
#define BSW_HALVES (NCHUNK*2*32*16)  // 13312 pre-swizzled B halves

static __device__ __forceinline__ unsigned short f32_to_bf16_bits(float f) {
    unsigned int u = __float_as_uint(f);
    unsigned int r = u + 0x7FFFu + ((u >> 16) & 1u);  // round-to-nearest-even
    return (unsigned short)(r >> 16);
}

// ---- kernel 0: build pre-swizzled bf16 B operands ----------------------
// Semantic K permutation (shared with A gather):
//   ci   = (K & 7) | ((K >> 4) << 3)
//   kpos = 2*kk + ((K >> 3) & 1)        (kpos == 25 -> zero pad)
// B VGPR layout (32x16, 16-bit): lane -> N = lane&15 (+16*tile),
//   half h -> K = (h&7) + 8*(lane>>4) + 16*(h>>3).
// Stored as [kk][tile][lane][16 halves] so a wave loads B with 2x b128.
__global__ void k_build_B(const float* __restrict__ w2,
                          unsigned short* __restrict__ bsw) {
    int idx = blockIdx.x * blockDim.x + threadIdx.x;
    if (idx >= BSW_HALVES) return;
    int h    = idx & 15;
    int lane = (idx >> 4) & 31;
    int tile = (idx >> 9) & 1;
    int kk   = idx >> 10;
    int K    = (h & 7) + ((lane >> 4) << 3) + ((h >> 3) << 4);
    int ci   = (K & 7) | ((K >> 4) << 3);
    int kpos = 2 * kk + ((K >> 3) & 1);
    unsigned short v = 0;
    if (kpos < 25) {
        int n = (lane & 15) + 16 * tile;   // output channel
        v = f32_to_bf16_bits(w2[(n * C1 + ci) * 25 + kpos]);
    }
    bsw[idx] = v;
}

// ---- kernel 1: conv1 + relu -> channel-last bf16 activations -----------
// act1 layout: [b][y][x][ci], 16 bf16 = 32 contiguous (aligned) bytes per
// spatial position.  One thread per position computes all 16 channels;
// w1/b1 addresses are wave-uniform -> scalar loads.
__global__ void __launch_bounds__(256)
k_conv1_relu(const float* __restrict__ x, const float* __restrict__ w1,
             const float* __restrict__ b1, unsigned short* __restrict__ act1) {
    int idx = blockIdx.x * blockDim.x + threadIdx.x;   // b*3600 + yy*60 + xx
    const int total = B_ * H1 * H1;
    if (idx >= total) return;
    int xx = idx % H1;
    int t  = idx / H1;
    int yy = t % H1;
    int b  = t / H1;
    const float* xin = x + (size_t)b * H_IN * H_IN + yy * H_IN + xx;

    float win[25];
    #pragma unroll
    for (int ky = 0; ky < 5; ++ky)
        #pragma unroll
        for (int kx = 0; kx < 5; ++kx)
            win[ky * 5 + kx] = xin[ky * H_IN + kx];

    union { unsigned short u[16]; uint4 q[2]; } o;
    #pragma unroll
    for (int c = 0; c < C1; ++c) {
        float acc = b1[c];
        #pragma unroll
        for (int k = 0; k < 25; ++k)
            acc = fmaf(win[k], w1[c * 25 + k], acc);
        o.u[c] = f32_to_bf16_bits(fmaxf(acc, 0.0f));
    }
    uint4* dst = (uint4*)(act1 + (size_t)idx * C1);
    dst[0] = o.q[0];
    dst[1] = o.q[1];
}

// ---- kernel 2: conv2 as WMMA GEMM + bias + relu + 2x2 maxpool ----------
// Per wave: 16 flattened output pixels x 32 channels, K fully unrolled.
// With the K permutation, A half h == channel h at kpos = 2*kk + hi, and
// channel-last act1 makes each chunk's A operand 32 contiguous bytes:
// two aligned b128 loads at compile-time offsets off a per-lane base.
__global__ void __launch_bounds__(256)
k_conv2_wmma(const unsigned short* __restrict__ act1,
             const unsigned short* __restrict__ bsw,
             const float* __restrict__ b2,
             unsigned int* __restrict__ cur_bits) {
    const int lane = threadIdx.x & 31;
    const int wave = threadIdx.x >> 5;
    const int tile = blockIdx.x * 8 + wave;
    const int b  = tile / TILES_PER_IMG;
    const int pt = tile % TILES_PER_IMG;
    const int p0 = pt * 16;

    const int mrow = lane & 15;       // A: M row / B,D: N col
    const int hi   = lane >> 4;
    const int p = p0 + mrow;
    const int y = p / H2;
    const int x = p % H2;

    const unsigned short* a0 = act1 + ((size_t)b * H1 * H1 + y * H1 + x) * C1;
    const unsigned short* aP1  = a0 + hi * C1;        // chunks with kx(2kk) < 4
    const unsigned short* aP56 = a0 + hi * (56 * C1); // chunks with kx(2kk) == 4
    const unsigned short* bb = bsw + lane * 16;

    v8f c0 = {}, c1 = {};

    #pragma unroll
    for (int kk = 0; kk < NCHUNK; ++kk) {
        const int kpos0 = 2 * kk;             // folded to constants
        const int ky0 = kpos0 / 5;
        const int kx0 = kpos0 % 5;
        const int sp  = (ky0 * 60 + kx0) * C1;  // elem offset, hi==0
        const unsigned short* ab = (kx0 < 4) ? aP1 : aP56;

        union { uint4 q[2]; v16bf v; } A;
        if (kk == NCHUNK - 1) {
            // kpos = 24 for hi==0 lanes; hi==1 lanes are K-padding -> 0.
            A.q[0] = uint4{0, 0, 0, 0};
            A.q[1] = uint4{0, 0, 0, 0};
            if (!hi) {
                A.q[0] = *(const uint4*)(ab + sp);
                A.q[1] = *(const uint4*)(ab + sp + 8);
            }
        } else {
            A.q[0] = *(const uint4*)(ab + sp);
            A.q[1] = *(const uint4*)(ab + sp + 8);
        }

        union { uint4 q[2]; v16bf v; } B0, B1;
        B0.q[0] = *(const uint4*)(bb + (kk * 2 + 0) * 512);
        B0.q[1] = *(const uint4*)(bb + (kk * 2 + 0) * 512 + 8);
        B1.q[0] = *(const uint4*)(bb + (kk * 2 + 1) * 512);
        B1.q[1] = *(const uint4*)(bb + (kk * 2 + 1) * 512 + 8);

        c0 = __builtin_amdgcn_wmma_f32_16x16x32_bf16(false, A.v, false, B0.v,
                                                     (short)0, c0, false, false);
        c1 = __builtin_amdgcn_wmma_f32_16x16x32_bf16(false, A.v, false, B1.v,
                                                     (short)0, c1, false, false);
    }

    // D layout: N = lane&15, M = r + 8*hi.  bias + relu, then 2x2 maxpool
    // via atomic-umax on IEEE bits (values >= 0 after relu).
    unsigned int* curb = cur_bits + (size_t)b * NFEAT;
    float bias0 = b2[mrow];
    float bias1 = b2[mrow + 16];
    #pragma unroll
    for (int r = 0; r < 8; ++r) {
        int m  = r + (hi << 3);
        int pp = p0 + m;
        int yy = pp / H2, xx = pp % H2;
        int py = yy >> 1, px = xx >> 1;
        float v0 = fmaxf(c0[r] + bias0, 0.0f);
        float v1 = fmaxf(c1[r] + bias1, 0.0f);
        atomicMax(&curb[ mrow       * (HP * HP) + py * HP + px], __float_as_uint(v0));
        atomicMax(&curb[(mrow + 16) * (HP * HP) + py * HP + px], __float_as_uint(v1));
    }
}

// ---- kernel 3: out[t,b,j] = bf[j] (atomic-add target init) -------------
__global__ void k_init_out(float* __restrict__ out, const float* __restrict__ bf) {
    int idx = blockIdx.x * blockDim.x + threadIdx.x;
    if (idx < NSTEPS * B_ * NOUT) out[idx] = bf[idx % NOUT];
}

// ---- kernel 4: fused 25-step LIF + readout -----------------------------
__global__ void __launch_bounds__(256)
k_lif(const float* __restrict__ cur, const float* __restrict__ wf,
      float* __restrict__ out) {
    const int CHUNKS = 13;  // ceil(25088 / 2048)
    int b     = blockIdx.x / CHUNKS;
    int chunk = blockIdx.x % CHUNKS;
    int tid   = threadIdx.x;
    int i0    = chunk * 2048 + tid * 8;
    bool active = (i0 < NFEAT);   // 25088 % 8 == 0, whole-thread guard

    float curv[8], mem[8], wfr[NOUT][8];
    #pragma unroll
    for (int e = 0; e < 8; ++e) { mem[e] = 0.f; curv[e] = 0.f; }
    #pragma unroll
    for (int j = 0; j < NOUT; ++j)
        #pragma unroll
        for (int e = 0; e < 8; ++e) wfr[j][e] = 0.f;

    if (active) {
        const float* cp = cur + (size_t)b * NFEAT + i0;
        #pragma unroll
        for (int e = 0; e < 8; ++e) curv[e] = cp[e];
        #pragma unroll
        for (int j = 0; j < NOUT; ++j) {
            const float* wp = wf + (size_t)j * NFEAT + i0;
            #pragma unroll
            for (int e = 0; e < 8; ++e) wfr[j][e] = wp[e];
        }
    }

    __shared__ float acc[NOUT];

    #pragma unroll 1
    for (int t = 0; t < NSTEPS; ++t) {
        if (tid < NOUT) acc[tid] = 0.f;
        __syncthreads();

        float s[NOUT];
        #pragma unroll
        for (int j = 0; j < NOUT; ++j) s[j] = 0.f;

        #pragma unroll
        for (int e = 0; e < 8; ++e) {
            float reset = (mem[e] > 1.0f) ? 1.0f : 0.0f;   // from previous mem
            mem[e] = 0.9f * mem[e] + curv[e] - reset;       // BETA, THRESH=1
            float spk = (mem[e] > 1.0f) ? 1.0f : 0.0f;
            #pragma unroll
            for (int j = 0; j < NOUT; ++j) s[j] = fmaf(spk, wfr[j][e], s[j]);
        }

        #pragma unroll
        for (int j = 0; j < NOUT; ++j) {
            float v = s[j];
            v += __shfl_down(v, 16, 32);
            v += __shfl_down(v, 8, 32);
            v += __shfl_down(v, 4, 32);
            v += __shfl_down(v, 2, 32);
            v += __shfl_down(v, 1, 32);
            if ((tid & 31) == 0) atomicAdd(&acc[j], v);
        }
        __syncthreads();
        if (tid < NOUT)
            atomicAdd(&out[((size_t)t * B_ + b) * NOUT + tid], acc[tid]);
    }
}

// ---- launcher ----------------------------------------------------------
extern "C" void kernel_launch(void* const* d_in, const int* in_sizes, int n_in,
                              void* d_out, int out_size, void* d_ws, size_t ws_size,
                              hipStream_t stream) {
    const float* x  = (const float*)d_in[0];
    const float* w1 = (const float*)d_in[1];
    const float* b1 = (const float*)d_in[2];
    const float* w2 = (const float*)d_in[3];
    const float* b2 = (const float*)d_in[4];
    const float* wf = (const float*)d_in[5];
    const float* bf = (const float*)d_in[6];
    float* out = (float*)d_out;

    char* ws = (char*)d_ws;
    size_t act1_bytes = (size_t)B_ * C1 * H1 * H1 * sizeof(unsigned short); // 118 MB
    size_t cur_bytes  = (size_t)B_ * NFEAT * sizeof(float);                 // 103 MB
    unsigned short* act1 = (unsigned short*)ws;
    float*          cur  = (float*)(ws + act1_bytes + 4096);                // pad
    unsigned short* bsw  = (unsigned short*)(ws + act1_bytes + 4096 + cur_bytes);

    k_build_B<<<(BSW_HALVES + 255) / 256, 256, 0, stream>>>(w2, bsw);

    int conv1_total = B_ * H1 * H1;
    k_conv1_relu<<<(conv1_total + 255) / 256, 256, 0, stream>>>(x, w1, b1, act1);

    hipMemsetAsync(cur, 0, cur_bytes, stream);   // maxpool targets = 0.0f bits
    k_init_out<<<(NSTEPS * B_ * NOUT + 255) / 256, 256, 0, stream>>>(out, bf);

    k_conv2_wmma<<<(B_ * TILES_PER_IMG) / 8, 256, 0, stream>>>(
        act1, bsw, b2, (unsigned int*)cur);

    k_lif<<<B_ * 13, 256, 0, stream>>>(cur, wf, out);
}
